// Reg_3stage_40931038331260
// MI455X (gfx1250) — matrix operand order repaired and checked by reading, not compiled, not run on hardware
//
#include <hip/hip_runtime.h>
#include <hip/hip_bf16.h>

// ---------------------------------------------------------------------------
// Types for CDNA5 WMMA (wave32)
// ---------------------------------------------------------------------------
typedef __attribute__((ext_vector_type(16))) __bf16 v16bf;
typedef __attribute__((ext_vector_type(8)))  __bf16 v8bf;
typedef __attribute__((ext_vector_type(8)))  float  v8f;

#define NPIX (320 * 480)        // 153600 pixels
#define LST  136                // LDS row stride in bf16 elems (128 + 8 pad, 16B aligned)

// packed-weight offsets (bf16 elements) inside workspace
constexpr long OFF_BB1 = 0;          // 128x128 -> 16384
constexpr long OFF_BB2 = 16384;
constexpr long OFF_BB3 = 32768;
constexpr long OFF_C1A = 49152;      // 32x128  -> 4096
constexpr long OFF_C1B = 53248;      // 32x32   -> 1024
constexpr long OFF_C1C = 54272;      // 16x32   -> 512
constexpr long OFF_C2A = 54784;
constexpr long OFF_C2B = 58880;
constexpr long OFF_C3A = 59904;
constexpr long OFF_C3B = 64000;
constexpr long OFF_R1  = 65024;      // 128x128 -> 16384
constexpr long OFF_W2  = 81408;      // 128x256 -> 32768 ; end 114176 elems
constexpr size_t OFF_INDS_BYTES = 229376;  // int32 inds[NPIX]

// ---------------------------------------------------------------------------
// Helpers
// ---------------------------------------------------------------------------
__device__ inline v8f zero8() {
  v8f z;
#pragma unroll
  for (int i = 0; i < 8; ++i) z[i] = 0.0f;
  return z;
}

__device__ inline v8f wmma_bf16(v16bf a, v16bf b, v8f c) {
  return __builtin_amdgcn_wmma_f32_16x16x32_bf16(false, a, false, b, (short)0, c,
                                                 false, false);
}

// A fragment (16x32 bf16, M=pixels, K=channels) from LDS row-major tile.
// Contiguous K order (matches packed B): lanes 0-15: K 0-15; lanes 16-31: K 16-31.
__device__ inline v16bf load_A(const __bf16* lds, int stride, int m0, int k0) {
  const int lane = threadIdx.x & 31;
  const int M = m0 + (lane & 15);
  const int kb = (lane & 16) ? 16 : 0;
  const __bf16* p = lds + M * stride + k0 + kb;      // 16B aligned
  v8bf lo = *(const v8bf*)p;                         // K = kb .. kb+7
  v8bf hi = *(const v8bf*)(p + 8);                   // K = kb+8 .. kb+15
  v16bf a;
#pragma unroll
  for (int j = 0; j < 8; ++j) { a[j] = lo[j]; a[j + 8] = hi[j]; }
  return a;
}

// B fragment from pre-packed global weights: one contiguous 32B blob per lane.
__device__ inline v16bf load_Bp(const __bf16* pack, int ntiles, int kt, int nt) {
  const int lane = threadIdx.x & 31;
  const __bf16* p = pack + ((((long)kt * ntiles + nt) * 32 + lane) << 4);
  return *(const v16bf*)p;
}

// Store D (16x16 f32) + bias + leaky_relu -> bf16 activation tile in LDS.
__device__ inline void store_act(__bf16* lds, int stride, int m0, int c0,
                                 v8f acc, const float* __restrict__ bias, int bc0) {
  const int lane = threadIdx.x & 31;
  const int N = lane & 15;
  const int mh = (lane & 16) ? 8 : 0;
  const float bv = bias[bc0 + N];
#pragma unroll
  for (int v = 0; v < 8; ++v) {
    float x = acc[v] + bv;
    x = (x >= 0.0f) ? x : 0.01f * x;
    lds[(m0 + mh + v) * stride + c0 + N] = (__bf16)x;
  }
}

// Store D + bias (optional leaky_relu) -> fp32 LDS tile.
__device__ inline void store_f32(float* lds, int stride, int m0, int c0,
                                 v8f acc, const float* __restrict__ bias, int bc0,
                                 bool relu) {
  const int lane = threadIdx.x & 31;
  const int N = lane & 15;
  const int mh = (lane & 16) ? 8 : 0;
  const float bv = bias[bc0 + N];
#pragma unroll
  for (int v = 0; v < 8; ++v) {
    float x = acc[v] + bv;
    if (relu) x = (x >= 0.0f) ? x : 0.01f * x;
    lds[(m0 + mh + v) * stride + c0 + N] = x;
  }
}

// ---------------------------------------------------------------------------
// Weight packing: fp32 [O][C] -> bf16 B-fragment layout (K=C, N=O)
// ---------------------------------------------------------------------------
__global__ void pack_dense_bf16(const float* __restrict__ W, int O, int C,
                                __bf16* __restrict__ dst) {
  const int e = blockIdx.x * 256 + threadIdx.x;
  if (e >= O * C) return;
  const int j = e & 15;
  const int lane = (e >> 4) & 31;
  const int tile = e >> 9;
  const int ntiles = O >> 4;
  const int nt = tile % ntiles;
  const int kt = tile / ntiles;
  const int n = nt * 16 + (lane & 15);
  const int k = kt * 32 + ((lane & 16) ? 16 : 0) + j;
  dst[e] = (__bf16)W[n * C + k];
}

// W2 [8][128][32] -> B layout with N = 32*k + o (K=128, N=256)
__global__ void pack_w2_bf16(const float* __restrict__ W2, __bf16* __restrict__ dst) {
  const int e = blockIdx.x * 256 + threadIdx.x;
  if (e >= 128 * 256) return;
  const int j = e & 15;
  const int lane = (e >> 4) & 31;
  const int tile = e >> 9;
  const int nt = tile % 16;
  const int kt = tile / 16;
  const int n = nt * 16 + (lane & 15);
  const int k = kt * 32 + ((lane & 16) ? 16 : 0) + j;
  const int kk = n >> 5, o = n & 31;
  dst[e] = (__bf16)W2[(kk * 128 + k) * 32 + o];
}

// ---------------------------------------------------------------------------
// GEMM building blocks (per-wave)
// ---------------------------------------------------------------------------
// 64px x 128out, K=128. Wave w handles nt {2w,2w+1}, all 4 M-tiles.
__device__ void gemm128_layer(const __bf16* in, __bf16* out, const __bf16* wp,
                              const float* __restrict__ bias, int w) {
  v8f acc[4][2];
#pragma unroll
  for (int mt = 0; mt < 4; ++mt) { acc[mt][0] = zero8(); acc[mt][1] = zero8(); }
#pragma unroll
  for (int kt = 0; kt < 4; ++kt) {
    v16bf b0 = load_Bp(wp, 8, kt, 2 * w);
    v16bf b1 = load_Bp(wp, 8, kt, 2 * w + 1);
#pragma unroll
    for (int mt = 0; mt < 4; ++mt) {
      v16bf a = load_A(in, LST, mt * 16, kt * 32);
      acc[mt][0] = wmma_bf16(a, b0, acc[mt][0]);
      acc[mt][1] = wmma_bf16(a, b1, acc[mt][1]);
    }
  }
#pragma unroll
  for (int mt = 0; mt < 4; ++mt)
#pragma unroll
    for (int j = 0; j < 2; ++j)
      store_act(out, LST, mt * 16, (2 * w + j) * 16, acc[mt][j], bias, (2 * w + j) * 16);
}

// 128 -> 32 head; wave w owns its own pixel rows [16w,16w+16)
__device__ void head128to32(const __bf16* in, __bf16* out, int cdst,
                            const __bf16* wp, const float* __restrict__ bias, int w) {
  v8f a0 = zero8(), a1 = zero8();
#pragma unroll
  for (int kt = 0; kt < 4; ++kt) {
    v16bf b0 = load_Bp(wp, 2, kt, 0);
    v16bf b1 = load_Bp(wp, 2, kt, 1);
    v16bf a = load_A(in, LST, w * 16, kt * 32);
    a0 = wmma_bf16(a, b0, a0);
    a1 = wmma_bf16(a, b1, a1);
  }
  store_act(out, LST, w * 16, cdst, a0, bias, 0);
  store_act(out, LST, w * 16, cdst + 16, a1, bias, 16);
}

__device__ void head32to32_act(const __bf16* in, int csrc, __bf16* out, int cdst,
                               const __bf16* wp, const float* __restrict__ bias, int w) {
  v16bf a = load_A(in, LST, w * 16, csrc);
  v16bf b0 = load_Bp(wp, 2, 0, 0);
  v16bf b1 = load_Bp(wp, 2, 0, 1);
  v8f a0 = wmma_bf16(a, b0, zero8());
  v8f a1 = wmma_bf16(a, b1, zero8());
  store_act(out, LST, w * 16, cdst, a0, bias, 0);
  store_act(out, LST, w * 16, cdst + 16, a1, bias, 16);
}

__device__ void head32to32_f32(const __bf16* in, int csrc, float* out, int ostride,
                               const __bf16* wp, const float* __restrict__ bias, int w,
                               bool relu) {
  v16bf a = load_A(in, LST, w * 16, csrc);
  v16bf b0 = load_Bp(wp, 2, 0, 0);
  v16bf b1 = load_Bp(wp, 2, 0, 1);
  v8f a0 = wmma_bf16(a, b0, zero8());
  v8f a1 = wmma_bf16(a, b1, zero8());
  store_f32(out, ostride, w * 16, 0, a0, bias, 0, relu);
  store_f32(out, ostride, w * 16, 16, a1, bias, 16, relu);
}

// ---------------------------------------------------------------------------
// Kernel 1: mask head (tiny, scalar, coalesced over pixels)
// ---------------------------------------------------------------------------
__global__ __launch_bounds__(256) void mask_kernel(
    const float* __restrict__ x_in,
    const float* __restrict__ w1, const float* __restrict__ b1,
    const float* __restrict__ w2, const float* __restrict__ b2,
    const float* __restrict__ w3, const float* __restrict__ b3,
    float* __restrict__ mout) {
  const int n = blockIdx.x * 256 + threadIdx.x;
  if (n >= NPIX) return;
  float m1[32];
#pragma unroll
  for (int o = 0; o < 32; ++o) m1[o] = b1[o];
  for (int c = 0; c < 128; ++c) {
    const float xv = x_in[c * NPIX + n];
#pragma unroll
    for (int o = 0; o < 32; ++o) m1[o] = fmaf(w1[o * 128 + c], xv, m1[o]);
  }
#pragma unroll
  for (int o = 0; o < 32; ++o) m1[o] = (m1[o] >= 0.f) ? m1[o] : 0.01f * m1[o];
  float m2[16];
#pragma unroll
  for (int o = 0; o < 16; ++o) m2[o] = b2[o];
  for (int i = 0; i < 32; ++i) {
#pragma unroll
    for (int o = 0; o < 16; ++o) m2[o] = fmaf(w2[o * 32 + i], m1[i], m2[o]);
  }
#pragma unroll
  for (int o = 0; o < 16; ++o) m2[o] = (m2[o] >= 0.f) ? m2[o] : 0.01f * m2[o];
  float mk = b3[0];
#pragma unroll
  for (int i = 0; i < 16; ++i) mk = fmaf(w3[i], m2[i], mk);
  mk = (mk >= 0.f) ? mk : 0.01f * mk;
  mout[n] = mk;
}

// ---------------------------------------------------------------------------
// Kernel 2: backbone + hierarchical classification -> inds[N]
// 64 pixels per workgroup, 4 waves (128 threads)
// ---------------------------------------------------------------------------
__global__ __launch_bounds__(128) void classify_kernel(
    const float* __restrict__ x_in,
    const float* __restrict__ bb1_b, const float* __restrict__ bb2_b,
    const float* __restrict__ bb3_b,
    const float* __restrict__ c1a_b, const float* __restrict__ c1b_b,
    const float* __restrict__ c1c_b,
    const float* __restrict__ c2a_b, const float* __restrict__ c2b_b,
    const float* __restrict__ c3a_b, const float* __restrict__ c3b_b,
    const float* __restrict__ Wc2, const float* __restrict__ bc2,
    const float* __restrict__ Wc3, const float* __restrict__ bc3,
    const __bf16* __restrict__ pk,
    int* __restrict__ inds_out) {
  __shared__ __bf16 bufA[64 * LST];   // 17408 B
  __shared__ __bf16 bufB[64 * LST];   // 17408 B
  __shared__ float  l1f[64 * 17];     //  4352 B
  __shared__ float  f2f[64 * 33];     //  8448 B
  __shared__ float  f3f[64 * 33];     //  8448 B  (total 56064 B)

  const int tid = threadIdx.x;
  const int w = tid >> 5;
  const int n0 = blockIdx.x * 64;

  // x tile -> LDS bf16 [pixel][chan], coalesced on pixels
#pragma unroll 4
  for (int it = 0; it < 64; ++it) {
    const int e = it * 128 + tid;
    const int c = e >> 6, p = e & 63;
    bufA[p * LST + c] = (__bf16)x_in[c * NPIX + n0 + p];
  }
  __syncthreads();

  // backbone: x -> y1 -> y2 -> x_l (ends in bufB)
  gemm128_layer(bufA, bufB, pk + OFF_BB1, bb1_b, w); __syncthreads();
  gemm128_layer(bufB, bufA, pk + OFF_BB2, bb2_b, w); __syncthreads();
  gemm128_layer(bufA, bufB, pk + OFF_BB3, bb3_b, w); __syncthreads();

  // heads (bufA reused as bf16 scratch; each wave owns its 16 pixel rows)
  head128to32(bufB, bufA, 0, pk + OFF_C1A, c1a_b, w);           // a1 -> cols 0..31
  head32to32_act(bufA, 0, bufA, 32, pk + OFF_C1B, c1b_b, w);    // b1 -> cols 32..63
  {                                                             // l1 = c1c * b1
    v16bf a = load_A(bufA, LST, w * 16, 32);
    v16bf b = load_Bp(pk + OFF_C1C, 1, 0, 0);
    v8f acc = wmma_bf16(a, b, zero8());
    store_f32(l1f, 17, w * 16, 0, acc, c1c_b, 0, false);
  }
  head128to32(bufB, bufA, 64, pk + OFF_C2A, c2a_b, w);          // a2 -> cols 64..95
  head32to32_f32(bufA, 64, f2f, 33, pk + OFF_C2B, c2b_b, w, true);
  head128to32(bufB, bufA, 96, pk + OFF_C3A, c3a_b, w);          // a3 -> cols 96..127
  head32to32_f32(bufA, 96, f3f, 33, pk + OFF_C3B, c3b_b, w, true);
  __syncthreads();

  // scalar per-pixel hierarchical decision
  if (tid < 64) {
    const int p = tid;
    // argmax over 16 stage-1 logits
    float best = l1f[p * 17];
    int i1 = 0;
    for (int j = 1; j < 16; ++j) {
      const float v = l1f[p * 17 + j];
      if (v > best) { best = v; i1 = j; }
    }
    float fr[32];
#pragma unroll
    for (int i = 0; i < 32; ++i) fr[i] = f2f[p * 33 + i];
    // stage 2: gathered CondMul(16,32,32)
    const float* wc = Wc2 + i1 * 1024;
    const float* bcv = bc2 + i1 * 32;
    float bst = -3.4e38f; int o2 = 0;
    for (int o = 0; o < 32; ++o) {
      float s = bcv[o];
      for (int i = 0; i < 32; ++i) s = fmaf(fr[i], wc[i * 32 + o], s);
      if (s > bst) { bst = s; o2 = o; }
    }
    int i12 = i1 * 16 + o2 - 8;
    i12 = i12 < 0 ? 0 : (i12 > 255 ? 255 : i12);
    // stage 3: gathered CondMul(256,32,32)
#pragma unroll
    for (int i = 0; i < 32; ++i) fr[i] = f3f[p * 33 + i];
    wc = Wc3 + i12 * 1024;
    bcv = bc3 + i12 * 32;
    bst = -3.4e38f; int o3 = 0;
    for (int o = 0; o < 32; ++o) {
      float s = bcv[o];
      for (int i = 0; i < 32; ++i) s = fmaf(fr[i], wc[i * 32 + o], s);
      if (s > bst) { bst = s; o3 = o; }
    }
    int idx = i12 * 16 + o3 - 8;
    idx = idx < 0 ? 0 : (idx > 4095 ? 4095 : idx);
    inds_out[n0 + p] = idx;
  }
}

// ---------------------------------------------------------------------------
// Kernel 3: regression head. 32 pixels per workgroup, 4 waves.
// ---------------------------------------------------------------------------
__global__ __launch_bounds__(128) void regress_kernel(
    const float* __restrict__ x_in,
    const float* __restrict__ r1_b,
    const float* __restrict__ b2, const float* __restrict__ W3,
    const float* __restrict__ b3,
    const __bf16* __restrict__ pk,
    const int* __restrict__ inds,
    float* __restrict__ out) {
  __shared__ __bf16 bufA[32 * LST];     //  8704 B
  __shared__ __bf16 bufB[32 * LST];     //  8704 B
  __shared__ float  all8[32 * 257];     // 32896 B  (total 50304 B)

  const int tid = threadIdx.x;
  const int w = tid >> 5;
  const int n0 = blockIdx.x * 32;

#pragma unroll 4
  for (int it = 0; it < 32; ++it) {
    const int e = it * 128 + tid;
    const int c = e >> 5, p = e & 31;
    bufA[p * LST + c] = (__bf16)x_in[c * NPIX + n0 + p];
  }
  __syncthreads();

  // xr = lrelu(r1 * x + b): 2 mt x 8 nt tiles; wave w: nt {2w,2w+1}
  {
    v8f acc[2][2];
#pragma unroll
    for (int mt = 0; mt < 2; ++mt) { acc[mt][0] = zero8(); acc[mt][1] = zero8(); }
#pragma unroll
    for (int kt = 0; kt < 4; ++kt) {
      v16bf b0 = load_Bp(pk + OFF_R1, 8, kt, 2 * w);
      v16bf b1 = load_Bp(pk + OFF_R1, 8, kt, 2 * w + 1);
#pragma unroll
      for (int mt = 0; mt < 2; ++mt) {
        v16bf a = load_A(bufA, LST, mt * 16, kt * 32);
        acc[mt][0] = wmma_bf16(a, b0, acc[mt][0]);
        acc[mt][1] = wmma_bf16(a, b1, acc[mt][1]);
      }
    }
#pragma unroll
    for (int mt = 0; mt < 2; ++mt)
#pragma unroll
      for (int j = 0; j < 2; ++j)
        store_act(bufB, LST, mt * 16, (2 * w + j) * 16, acc[mt][j], r1_b,
                  (2 * w + j) * 16);
  }
  __syncthreads();

  // all8[p, k*32+o] = xr . W2[k] + b2[k]  (dense over all 8 super-classes)
  {
    v8f acc[2][4];
#pragma unroll
    for (int mt = 0; mt < 2; ++mt)
#pragma unroll
      for (int q = 0; q < 4; ++q) acc[mt][q] = zero8();
#pragma unroll
    for (int kt = 0; kt < 4; ++kt) {
      v16bf bv[4];
#pragma unroll
      for (int q = 0; q < 4; ++q) bv[q] = load_Bp(pk + OFF_W2, 16, kt, 4 * w + q);
#pragma unroll
      for (int mt = 0; mt < 2; ++mt) {
        v16bf a = load_A(bufB, LST, mt * 16, kt * 32);
#pragma unroll
        for (int q = 0; q < 4; ++q) acc[mt][q] = wmma_bf16(a, bv[q], acc[mt][q]);
      }
    }
#pragma unroll
    for (int mt = 0; mt < 2; ++mt)
#pragma unroll
      for (int q = 0; q < 4; ++q)
        store_f32(all8, 257, mt * 16, (4 * w + q) * 16, acc[mt][q], b2,
                  (4 * w + q) * 16, false);
  }
  __syncthreads();

  if (tid < 32) {
    const int p = tid, n = n0 + p;
    const int idx = inds[n];
    const int s = idx >> 9;                       // inds / 512
    float r = b3[idx];
    const float* w3 = W3 + idx * 32;
    const float* arow = all8 + p * 257 + s * 32;
#pragma unroll 8
    for (int o = 0; o < 32; ++o) {
      float h = arow[o];
      h = (h >= 0.f) ? h : 0.01f * h;
      r = fmaf(h, w3[o], r);
    }
    out[n] = ((float)idx + r) * (1.0f / 4096.0f);
  }
}

// ---------------------------------------------------------------------------
// Launch
// ---------------------------------------------------------------------------
extern "C" void kernel_launch(void* const* d_in, const int* in_sizes, int n_in,
                              void* d_out, int out_size, void* d_ws, size_t ws_size,
                              hipStream_t stream) {
  const float* x_in   = (const float*)d_in[0];
  const float* bb1_w  = (const float*)d_in[1];
  const float* bb1_b  = (const float*)d_in[2];
  const float* bb2_w  = (const float*)d_in[3];
  const float* bb2_b  = (const float*)d_in[4];
  const float* bb3_w  = (const float*)d_in[5];
  const float* bb3_b  = (const float*)d_in[6];
  const float* msk1_w = (const float*)d_in[7];
  const float* msk1_b = (const float*)d_in[8];
  const float* msk2_w = (const float*)d_in[9];
  const float* msk2_b = (const float*)d_in[10];
  const float* msk3_w = (const float*)d_in[11];
  const float* msk3_b = (const float*)d_in[12];
  const float* c1a_w  = (const float*)d_in[13];
  const float* c1a_b  = (const float*)d_in[14];
  const float* c1b_w  = (const float*)d_in[15];
  const float* c1b_b  = (const float*)d_in[16];
  const float* c1c_w  = (const float*)d_in[17];
  const float* c1c_b  = (const float*)d_in[18];
  const float* c2a_w  = (const float*)d_in[19];
  const float* c2a_b  = (const float*)d_in[20];
  const float* c2b_w  = (const float*)d_in[21];
  const float* c2b_b  = (const float*)d_in[22];
  const float* Wc2    = (const float*)d_in[23];
  const float* bc2    = (const float*)d_in[24];
  const float* c3a_w  = (const float*)d_in[25];
  const float* c3a_b  = (const float*)d_in[26];
  const float* c3b_w  = (const float*)d_in[27];
  const float* c3b_b  = (const float*)d_in[28];
  const float* Wc3    = (const float*)d_in[29];
  const float* bc3    = (const float*)d_in[30];
  const float* r1_w   = (const float*)d_in[31];
  const float* r1_b   = (const float*)d_in[32];
  const float* W2     = (const float*)d_in[33];
  const float* b2     = (const float*)d_in[34];
  const float* W3     = (const float*)d_in[35];
  const float* b3     = (const float*)d_in[36];

  __bf16* wsb = (__bf16*)d_ws;
  int* inds = (int*)((char*)d_ws + OFF_INDS_BYTES);
  float* out = (float*)d_out;           // output 0: [N]
  float* mask = out + NPIX;             // output 1: [N]

  auto pack = [&](const float* Wm, int O, int C, long off) {
    const int total = O * C;
    pack_dense_bf16<<<(total + 255) / 256, 256, 0, stream>>>(Wm, O, C, wsb + off);
  };
  pack(bb1_w, 128, 128, OFF_BB1);
  pack(bb2_w, 128, 128, OFF_BB2);
  pack(bb3_w, 128, 128, OFF_BB3);
  pack(c1a_w, 32, 128, OFF_C1A);
  pack(c1b_w, 32, 32, OFF_C1B);
  pack(c1c_w, 16, 32, OFF_C1C);
  pack(c2a_w, 32, 128, OFF_C2A);
  pack(c2b_w, 32, 32, OFF_C2B);
  pack(c3a_w, 32, 128, OFF_C3A);
  pack(c3b_w, 32, 32, OFF_C3B);
  pack(r1_w, 128, 128, OFF_R1);
  pack_w2_bf16<<<128, 256, 0, stream>>>(W2, wsb + OFF_W2);

  mask_kernel<<<NPIX / 256, 256, 0, stream>>>(x_in, msk1_w, msk1_b, msk2_w, msk2_b,
                                              msk3_w, msk3_b, mask);

  classify_kernel<<<NPIX / 64, 128, 0, stream>>>(
      x_in, bb1_b, bb2_b, bb3_b, c1a_b, c1b_b, c1c_b, c2a_b, c2b_b, c3a_b, c3b_b,
      Wc2, bc2, Wc3, bc3, wsb, inds);

  regress_kernel<<<NPIX / 32, 128, 0, stream>>>(x_in, r1_b, b2, W3, b3, wsb, inds,
                                                out);
}
